// MultiHeadAttention_34067680591877
// MI455X (gfx1250) — matrix operand order repaired and checked by reading, not compiled, not run on hardware
//
#include <hip/hip_runtime.h>
#include <hip/hip_bf16.h>

// ---------------------------------------------------------------------------
// MHA forward for MI455X (gfx1250, wave32): bf16 WMMA 16x16x32, f32 accumulate
// ---------------------------------------------------------------------------

#define T_LEN 512
#define S_LEN 512
#define BATCH 32
#define EMB   512
#define ADIM  512
#define NHEAD 8
#define HDIM  64
#define MROWS (T_LEN * BATCH)   // 16384

typedef __bf16 bf16_t;
typedef bf16_t v16bf __attribute__((ext_vector_type(16)));
typedef float  v8f   __attribute__((ext_vector_type(8)));
typedef float  v4f   __attribute__((ext_vector_type(4)));
typedef unsigned int v4u __attribute__((ext_vector_type(4)));
typedef unsigned int v2u __attribute__((ext_vector_type(2)));

union FragB {
    v16bf v;
    v4u q[2];
    unsigned short s[16];
};

// Full conversion (round-to-nearest-even, preserves inf/nan) — pre-pass only.
__device__ __forceinline__ unsigned short f2bf(float f) {
    unsigned int u = __builtin_bit_cast(unsigned int, f);
    unsigned int r = u + 0x7FFFu + ((u >> 16) & 1u);
    if ((u & 0x7F800000u) == 0x7F800000u) r = u;
    return (unsigned short)(r >> 16);
}

// Fast conversion for known-finite values (RNE, no special-case guard).
__device__ __forceinline__ unsigned short f2bf_fast(float f) {
    unsigned int u = __builtin_bit_cast(unsigned int, f);
    return (unsigned short)((u + 0x7FFFu + ((u >> 16) & 1u)) >> 16);
}

__device__ __forceinline__ v8f wmma_bf16(const FragB& a, const FragB& b, v8f c) {
    return __builtin_amdgcn_wmma_f32_16x16x32_bf16(
        /*neg_a=*/false, a.v, /*neg_b=*/false, b.v,
        /*c_mod=*/(short)0, c, /*reuse_a=*/false, /*reuse_b=*/false);
}

// ---------------------------------------------------------------------------
// fp32 -> bf16, vectorized x4 (weights and activations pre-pass)
// ---------------------------------------------------------------------------
__global__ __launch_bounds__(256) void cvt4_kernel(const float* __restrict__ in,
                                                   unsigned short* __restrict__ out,
                                                   int n4) {
    int i = blockIdx.x * 256 + threadIdx.x;
    if (i < n4) {
        v4f f = ((const v4f*)in)[i];
        v2u p;
        p[0] = (unsigned)f2bf(f[0]) | ((unsigned)f2bf(f[1]) << 16);
        p[1] = (unsigned)f2bf(f[2]) | ((unsigned)f2bf(f[3]) << 16);
        ((v2u*)out)[i] = p;
    }
}

// ---------------------------------------------------------------------------
// NT GEMM:  C[m,n] = sum_k A[m,k] * W[n,k] + bias[n]    (bf16 x bf16 -> f32)
//   M = MROWS, N = 512, K = 512. Both operands K-contiguous.
//   STORE: 0 = bf16 natural (row*512+col)   (Q/K projections)
//          1 = bf16 V-transposed (B,H,D,S)  (V projection)
//          2 = f32  natural                 (output projection)
// Workgroup: 256 threads = 8 waves, tile 128(M) x 128(N); each wave 32x64
// (2 A-frags x 4 B-frags -> 8 WMMAs per K-step, B-frag reused twice).
// ---------------------------------------------------------------------------
template <int STORE>
__global__ __launch_bounds__(256) void gemm_nt_kernel(const unsigned short* __restrict__ A,
                                                      const unsigned short* __restrict__ W,
                                                      const float* __restrict__ bias,
                                                      void* __restrict__ Op) {
    const int lane = threadIdx.x & 31;
    const int wave = threadIdx.x >> 5;
    const int ln   = lane & 15;
    const int lh   = lane >> 4;
    const int m0 = blockIdx.x * 128 + (wave >> 1) * 32;
    const int n0 = blockIdx.y * 128 + (wave & 1) * 64;
    const int klo = lh * 8;      // 16-bit A layout: K chunks [klo..klo+7], [16+klo..+7]

    v8f acc[2][4] = {};

    for (int k0 = 0; k0 < 512; k0 += 32) {
        FragB a[2];
#pragma unroll
        for (int m = 0; m < 2; ++m) {
            const unsigned short* ap = A + (size_t)(m0 + m * 16 + ln) * 512 + k0;
            a[m].q[0] = *(const v4u*)(ap + klo);
            a[m].q[1] = *(const v4u*)(ap + 16 + klo);
            if (k0 + 32 < 512) __builtin_prefetch(ap + 32, 0, 2);
        }
#pragma unroll
        for (int f = 0; f < 4; ++f) {
            // B-operand: lane holds column n = lane&15; K chunks [(lane>>4)*8..+7], [16+(lane>>4)*8..+7]
            const unsigned short* wp = W + (size_t)(n0 + f * 16 + ln) * 512 + k0 + lh * 8;
            FragB bfr;
            bfr.q[0] = *(const v4u*)wp;
            bfr.q[1] = *(const v4u*)(wp + 16);
            acc[0][f] = wmma_bf16(a[0], bfr, acc[0][f]);
            acc[1][f] = wmma_bf16(a[1], bfr, acc[1][f]);
        }
    }

    // C/D layout: lane l, vgpr r -> row = r + 8*(l>>4), col = l&15 (per 16-col frag)
#pragma unroll
    for (int m = 0; m < 2; ++m) {
#pragma unroll
        for (int f = 0; f < 4; ++f) {
            const int col = n0 + f * 16 + ln;
            const float bv = bias[col];
#pragma unroll
            for (int r = 0; r < 8; ++r) {
                const int row = m0 + m * 16 + r + lh * 8;
                const float v = acc[m][f][r] + bv;
                if (STORE == 2) {
                    ((float*)Op)[(size_t)row * 512 + col] = v;
                } else if (STORE == 0) {
                    ((unsigned short*)Op)[(size_t)row * 512 + col] = f2bf_fast(v);
                } else {
                    // row = s*BATCH + b ; col = h*HDIM + d ; store (B,H,D,S)
                    const int s = row >> 5, b = row & 31;
                    const int h = col >> 6, d = col & 63;
                    ((unsigned short*)Op)[((size_t)(b * NHEAD + h) * HDIM + d) * S_LEN + s] =
                        f2bf_fast(v);
                }
            }
        }
    }
}

// ---------------------------------------------------------------------------
// Flash attention core: per (b,h) and 64-row t-tile. 128 threads = 4 waves,
// each wave owns 16 query rows. S processed in chunks of 32 (= WMMA K).
// ---------------------------------------------------------------------------
__global__ __launch_bounds__(128) void flash_kernel(const unsigned short* __restrict__ Qp,
                                                    const unsigned short* __restrict__ Kp,
                                                    const unsigned short* __restrict__ VpT,
                                                    const unsigned char* __restrict__ mask,
                                                    unsigned short* __restrict__ Aout) {
    __shared__ unsigned short plds[4 * 16 * 32];   // per-wave 16x32 bf16 P tile

    const int lane = threadIdx.x & 31;
    const int wave = threadIdx.x >> 5;
    const int ln = lane & 15;
    const int lh = lane >> 4;
    const int klo = lh * 8;

    const int bh = blockIdx.y;          // b*NHEAD + h
    const int b = bh >> 3;
    const int h = bh & 7;
    const int t0 = blockIdx.x * 64 + wave * 16;
    const int wbase = wave * 512;       // LDS element offset for this wave
    const float NEG_INF = -__builtin_inff();

    // Q A-fragments (loop invariant): K = 64 -> two kbase chunks {0,32}
    FragB aq[2];
    {
        const unsigned short* qp = Qp + ((size_t)(t0 + ln) * BATCH + b) * 512 + h * HDIM;
#pragma unroll
        for (int kb = 0; kb < 2; ++kb) {
            aq[kb].q[0] = *(const v4u*)(qp + kb * 32 + klo);
            aq[kb].q[1] = *(const v4u*)(qp + kb * 32 + 16 + klo);
        }
    }

    float m_run[8], l_run[8];
#pragma unroll
    for (int r = 0; r < 8; ++r) { m_run[r] = NEG_INF; l_run[r] = 0.0f; }
    v8f o[4] = {};

    for (int s0 = 0; s0 < S_LEN; s0 += 32) {
        // ---- scores: (16 t-rows) x (32 s-cols), K = HDIM = 64 ----
        v8f sc[2] = {};
#pragma unroll
        for (int nf = 0; nf < 2; ++nf) {
            const unsigned short* kp =
                Kp + ((size_t)(s0 + nf * 16 + ln) * BATCH + b) * 512 + h * HDIM + lh * 8;
#pragma unroll
            for (int kb = 0; kb < 2; ++kb) {
                FragB kf;
                kf.q[0] = *(const v4u*)(kp + kb * 32);
                kf.q[1] = *(const v4u*)(kp + kb * 32 + 16);
                sc[nf] = wmma_bf16(aq[kb], kf, sc[nf]);
            }
        }
        // scale 1/sqrt(D) with additive mask bias folded into one FMA per element
#pragma unroll
        for (int nf = 0; nf < 2; ++nf) {
            const float mb = (mask[b * S_LEN + s0 + nf * 16 + ln] != 0) ? NEG_INF : 0.0f;
#pragma unroll
            for (int r = 0; r < 8; ++r)
                sc[nf][r] = fmaf(sc[nf][r], 0.125f, mb);
        }
        // ---- online softmax; clamped running max makes exp(-inf - mnc) -> 0
        //      without any per-element guards ----
        float scale_r[8];
#pragma unroll
        for (int r = 0; r < 8; ++r) {
            float rm = fmaxf(sc[0][r], sc[1][r]);
#pragma unroll
            for (int off = 1; off < 16; off <<= 1) rm = fmaxf(rm, __shfl_xor(rm, off, 32));
            const float mn  = fmaxf(m_run[r], rm);
            const float mnc = fmaxf(mn, -1.0e38f);       // finite clamp
            const float scl = __expf(m_run[r] - mnc);    // 0 when m_run == -inf
            const float p0 = __expf(sc[0][r] - mnc);     // 0 when score == -inf
            const float p1 = __expf(sc[1][r] - mnc);
            sc[0][r] = p0;
            sc[1][r] = p1;
            float rs = p0 + p1;
#pragma unroll
            for (int off = 1; off < 16; off <<= 1) rs += __shfl_xor(rs, off, 32);
            l_run[r] = l_run[r] * scl + rs;
            m_run[r] = mn;
            scale_r[r] = scl;
        }
#pragma unroll
        for (int f = 0; f < 4; ++f)
#pragma unroll
            for (int r = 0; r < 8; ++r) o[f][r] *= scale_r[r];

        // ---- P: C/D-layout regs -> LDS -> A-layout fragment ----
#pragma unroll
        for (int nf = 0; nf < 2; ++nf)
#pragma unroll
            for (int r = 0; r < 8; ++r)
                plds[wbase + (r + lh * 8) * 32 + nf * 16 + ln] = f2bf_fast(sc[nf][r]);
        asm volatile("s_wait_dscnt 0" ::: "memory");   // intra-wave LDS RAW
        FragB pa;
        pa.q[0] = *(const v4u*)&plds[wbase + ln * 32 + klo];
        pa.q[1] = *(const v4u*)&plds[wbase + ln * 32 + 16 + klo];

        // ---- PV: (16 x 32) * (32 x 64) accumulate into o ----
#pragma unroll
        for (int f = 0; f < 4; ++f) {
            const unsigned short* vp =
                VpT + ((size_t)bh * HDIM + f * 16 + ln) * S_LEN + s0 + lh * 8;
            FragB vf;
            vf.q[0] = *(const v4u*)vp;
            vf.q[1] = *(const v4u*)(vp + 16);
            o[f] = wmma_bf16(pa, vf, o[f]);
        }
    }

    // ---- epilogue: normalize by l (one rcp per row), store bf16 (T,B,A) ----
#pragma unroll
    for (int r = 0; r < 8; ++r) {
        const int t = t0 + r + lh * 8;
        const float inv = 1.0f / l_run[r];
#pragma unroll
        for (int f = 0; f < 4; ++f) {
            const int col = h * HDIM + f * 16 + ln;
            Aout[((size_t)t * BATCH + b) * 512 + col] = f2bf_fast(o[f][r] * inv);
        }
    }
}

// ---------------------------------------------------------------------------
// launch
// ---------------------------------------------------------------------------
extern "C" void kernel_launch(void* const* d_in, const int* in_sizes, int n_in,
                              void* d_out, int out_size, void* d_ws, size_t ws_size,
                              hipStream_t stream) {
    const float* query = (const float*)d_in[0];
    const float* key   = (const float*)d_in[1];
    const float* value = (const float*)d_in[2];
    const unsigned char* attn_mask = (const unsigned char*)d_in[3];
    const float* Wq = (const float*)d_in[4];
    const float* bq = (const float*)d_in[5];
    const float* Wk = (const float*)d_in[6];
    const float* bk = (const float*)d_in[7];
    const float* Wv = (const float*)d_in[8];
    const float* bv = (const float*)d_in[9];
    const float* Wo = (const float*)d_in[10];
    const float* bo = (const float*)d_in[11];
    float* out = (float*)d_out;

    char* ws = (char*)d_ws;
    size_t off = 0;
    const size_t WSZ = (size_t)ADIM * EMB * 2;          // one weight matrix, bf16
    const size_t PSZ = (size_t)MROWS * ADIM * 2;        // one activation tensor, bf16
    unsigned short* wWq = (unsigned short*)(ws + off); off += WSZ;
    unsigned short* wWk = (unsigned short*)(ws + off); off += WSZ;
    unsigned short* wWv = (unsigned short*)(ws + off); off += WSZ;
    unsigned short* wWo = (unsigned short*)(ws + off); off += WSZ;
    unsigned short* qb  = (unsigned short*)(ws + off); off += PSZ;  // bf16 query in
    unsigned short* kb  = (unsigned short*)(ws + off); off += PSZ;  // bf16 key in
    unsigned short* vb  = (unsigned short*)(ws + off); off += PSZ;  // bf16 value in
    unsigned short* Qp  = (unsigned short*)(ws + off); off += PSZ;
    unsigned short* Kp  = (unsigned short*)(ws + off); off += PSZ;
    unsigned short* VpT = (unsigned short*)(ws + off); off += PSZ;
    unsigned short* Acc = qb;   // alias: qb is dead after the Q projection

    // 1) fp32 -> bf16 pre-pass (weights + activations)
    const int wn4 = (ADIM * EMB) / 4;          // 65536
    const int an4 = (MROWS * EMB) / 4;         // 2097152
    cvt4_kernel<<<(wn4 + 255) / 256, 256, 0, stream>>>(Wq, wWq, wn4);
    cvt4_kernel<<<(wn4 + 255) / 256, 256, 0, stream>>>(Wk, wWk, wn4);
    cvt4_kernel<<<(wn4 + 255) / 256, 256, 0, stream>>>(Wv, wWv, wn4);
    cvt4_kernel<<<(wn4 + 255) / 256, 256, 0, stream>>>(Wo, wWo, wn4);
    cvt4_kernel<<<(an4 + 255) / 256, 256, 0, stream>>>(query, qb, an4);
    cvt4_kernel<<<(an4 + 255) / 256, 256, 0, stream>>>(key,   kb, an4);
    cvt4_kernel<<<(an4 + 255) / 256, 256, 0, stream>>>(value, vb, an4);

    // 2) input projections (M=16384, N=512, K=512), WG tile 128x128
    dim3 ggrid(MROWS / 128, 512 / 128);
    gemm_nt_kernel<0><<<ggrid, 256, 0, stream>>>(qb, wWq, bq, Qp);
    gemm_nt_kernel<0><<<ggrid, 256, 0, stream>>>(kb, wWk, bk, Kp);
    gemm_nt_kernel<1><<<ggrid, 256, 0, stream>>>(vb, wWv, bv, VpT);

    // 3) attention core: grid (t-tiles, B*H)
    flash_kernel<<<dim3(T_LEN / 64, BATCH * NHEAD), 128, 0, stream>>>(Qp, Kp, VpT, attn_mask, Acc);

    // 4) output projection -> fp32 d_out
    gemm_nt_kernel<2><<<ggrid, 256, 0, stream>>>(Acc, wWo, bo, out);
}
